// LolcatsLinearAttention_28243704939085
// MI455X (gfx1250) — matrix operand order repaired and checked by reading, not compiled, not run on hardware
//
#include <hip/hip_runtime.h>
#include <hip/hip_bf16.h>

#define L_SEQ 1024
#define HID   2048
#define NH    32
#define HD    64
#define FD    64

typedef __attribute__((ext_vector_type(16))) __bf16 v16bf;
typedef __attribute__((ext_vector_type(8)))  __bf16 v8bf;
typedef __attribute__((ext_vector_type(8)))  float  v8f;
typedef __attribute__((ext_vector_type(4)))  int    v4i;

#if defined(__HIP_DEVICE_COMPILE__) && defined(__has_builtin)
#  if __has_builtin(__builtin_amdgcn_global_load_async_to_lds_b128)
#    define HAVE_ASYNC 1
#  endif
#endif
#ifndef HAVE_ASYNC
#  define HAVE_ASYNC 0
#endif

#if HAVE_ASYNC
// Copy 16 bytes global -> LDS per lane, asynchronously (ASYNCcnt-tracked).
// Builtin signature (from toolchain diagnostic): (v4i* src, v4i* dst, imm, imm)
static __device__ __forceinline__ void async_cp16(void* lds, const void* g) {
  __builtin_amdgcn_global_load_async_to_lds_b128((v4i*)g, (v4i*)lds, 0, 0);
}
static __device__ __forceinline__ void wait_async0() {
#if __has_builtin(__builtin_amdgcn_s_wait_asynccnt)
  __builtin_amdgcn_s_wait_asynccnt(0);
#else
  asm volatile("s_wait_asynccnt 0" ::: "memory");
#endif
}
#endif

static __device__ __forceinline__ v8f wmma_bf16(v16bf a, v16bf b, v8f c) {
  // D = A(16x32 bf16) x B(32x16 bf16) + C(16x16 f32)
  return __builtin_amdgcn_wmma_f32_16x16x32_bf16(false, a, false, b, (short)0, c,
                                                 false, false);
}

// ---------------------------------------------------------------------------
// Fragment loaders (wave32; contiguous A/B k order used here, not the ISA table order):
//  A 16x32 (16-bit): lane holds row M=L&15; half=L>>4;
//    element e -> K = e + 16*half
//  B 32x16 (16-bit): lane holds col N=L&15;  element e -> K = e + 16*half
//  C/D 16x16 (f32): lane holds col N=L&15; reg r -> row M = r + 8*half
// ---------------------------------------------------------------------------

static __device__ __forceinline__ v16bf load_a_from_f32(const float* __restrict__ src,
                                                        int ld, int row0, int k0) {
  const int lane = threadIdx.x & 31;
  const int m = lane & 15, half = lane >> 4;
  const float* p = src + (size_t)(row0 + m) * ld + k0 + half * 16;
  float f[16];
  *reinterpret_cast<float4*>(f + 0)  = *reinterpret_cast<const float4*>(p + 0);
  *reinterpret_cast<float4*>(f + 4)  = *reinterpret_cast<const float4*>(p + 4);
  *reinterpret_cast<float4*>(f + 8)  = *reinterpret_cast<const float4*>(p + 8);
  *reinterpret_cast<float4*>(f + 12) = *reinterpret_cast<const float4*>(p + 12);
  v16bf a;
#pragma unroll
  for (int e = 0; e < 16; ++e) a[e] = (__bf16)f[e];
  return a;
}

static __device__ __forceinline__ v16bf load_a_from_bf16(const __bf16* __restrict__ src,
                                                         int ld, int row0, int k0) {
  const int lane = threadIdx.x & 31;
  const int m = lane & 15, half = lane >> 4;
  const __bf16* p = src + (size_t)(row0 + m) * ld + k0 + half * 16;
  v8bf lo = *reinterpret_cast<const v8bf*>(p);
  v8bf hi = *reinterpret_cast<const v8bf*>(p + 8);
  return __builtin_shufflevector(lo, hi, 0, 1, 2, 3, 4, 5, 6, 7,
                                 8, 9, 10, 11, 12, 13, 14, 15);
}

// A fragment where source is [K x M] row-major bf16 (transposed source).
static __device__ __forceinline__ v16bf load_aT_bf16(const __bf16* src,
                                                     int ld, int row0, int k0) {
  const int lane = threadIdx.x & 31;
  const int m = lane & 15, half = lane >> 4;
  const __bf16* base = src + (size_t)k0 * ld + row0 + m;
  v16bf a;
#pragma unroll
  for (int e = 0; e < 16; ++e) {
    int kk = e + half * 16;
    a[e] = base[(size_t)kk * ld];
  }
  return a;
}

// B fragment of W^T where W is [N x K] row-major fp32.
static __device__ __forceinline__ v16bf load_b_wT_f32(const float* __restrict__ W,
                                                      int ld, int k0, int col0) {
  const int lane = threadIdx.x & 31;
  const int n = lane & 15, half = lane >> 4;
  const float* p = W + (size_t)(col0 + n) * ld + k0 + half * 16;
  float f[16];
#pragma unroll
  for (int i = 0; i < 4; ++i)
    *reinterpret_cast<float4*>(f + 4 * i) = *reinterpret_cast<const float4*>(p + 4 * i);
  v16bf b;
#pragma unroll
  for (int e = 0; e < 16; ++e) b[e] = (__bf16)f[e];
  return b;
}

// B fragment from [K x N] row-major fp32 (strided along K).
static __device__ __forceinline__ v16bf load_b_kn_f32(const float* __restrict__ src,
                                                      int ld, int k0, int col0) {
  const int lane = threadIdx.x & 31;
  const int n = lane & 15, half = lane >> 4;
  const float* p = src + (size_t)(k0 + half * 16) * ld + col0 + n;
  v16bf b;
#pragma unroll
  for (int e = 0; e < 16; ++e) b[e] = (__bf16)p[(size_t)e * ld];
  return b;
}

// B fragment from [N x K] row-major bf16 (K contiguous per lane).
static __device__ __forceinline__ v16bf load_b_nk_bf16(const __bf16* src,
                                                       int ld, int k0, int col0) {
  const int lane = threadIdx.x & 31;
  const int n = lane & 15, half = lane >> 4;
  const __bf16* p = src + (size_t)(col0 + n) * ld + k0 + half * 16;
  v8bf lo = *reinterpret_cast<const v8bf*>(p);
  v8bf hi = *reinterpret_cast<const v8bf*>(p + 8);
  return __builtin_shufflevector(lo, hi, 0, 1, 2, 3, 4, 5, 6, 7,
                                 8, 9, 10, 11, 12, 13, 14, 15);
}

// B fragment from [K x N] row-major bf16 (strided along K). LDS or global.
static __device__ __forceinline__ v16bf load_b_kn_bf16(const __bf16* src,
                                                       int ld, int k0, int col0) {
  const int lane = threadIdx.x & 31;
  const int n = lane & 15, half = lane >> 4;
  const __bf16* p = src + (size_t)(k0 + half * 16) * ld + col0 + n;
  v16bf b;
#pragma unroll
  for (int e = 0; e < 16; ++e) b[e] = p[(size_t)e * ld];
  return b;
}

// ---------------------------------------------------------------------------
// K0: f32 -> bf16 conversion (8 elements/thread).
// ---------------------------------------------------------------------------
__global__ __launch_bounds__(256)
void k_cvt_f32_bf16(const float* __restrict__ s, __bf16* __restrict__ d, int n8) {
  int i = blockIdx.x * blockDim.x + threadIdx.x;
  if (i >= n8) return;
  const float4* s4 = reinterpret_cast<const float4*>(s) + (size_t)i * 2;
  float4 x0 = s4[0], x1 = s4[1];
  v8bf o;
  o[0] = (__bf16)x0.x; o[1] = (__bf16)x0.y; o[2] = (__bf16)x0.z; o[3] = (__bf16)x0.w;
  o[4] = (__bf16)x1.x; o[5] = (__bf16)x1.y; o[6] = (__bf16)x1.z; o[7] = (__bf16)x1.w;
  reinterpret_cast<v8bf*>(d)[i] = o;
}

// ---------------------------------------------------------------------------
// K1 fast: q/k/v = hidden @ W^T with pre-converted bf16 operands.
// One wave per 64x64 tile: 16 WMMAs per K-step, 1:1 b128-load:wmma ratio.
// ---------------------------------------------------------------------------
__global__ __launch_bounds__(32)
void k_proj_qkv_bf16(const __bf16* __restrict__ hb,
                     const __bf16* __restrict__ wqb, const __bf16* __restrict__ wkb,
                     const __bf16* __restrict__ wvb,
                     __bf16* __restrict__ qw, __bf16* __restrict__ kw,
                     __bf16* __restrict__ vw) {
  const int tileL = blockIdx.x * 64;
  const int tileO = blockIdx.y * 64;
  const int z = blockIdx.z;
  const __bf16* W = (z == 0) ? wqb : (z == 1) ? wkb : wvb;
  __bf16* dst     = (z == 0) ? qw : (z == 1) ? kw : vw;
  const int lane = threadIdx.x & 31;
  const int n = lane & 15, half = lane >> 4;

  v8f c[4][4] = {};
  for (int k0 = 0; k0 < HID; k0 += 32) {
    v16bf a[4], b[4];
#pragma unroll
    for (int i = 0; i < 4; ++i) a[i] = load_a_from_bf16(hb, HID, tileL + 16 * i, k0);
#pragma unroll
    for (int j = 0; j < 4; ++j) b[j] = load_b_nk_bf16(W, HID, k0, tileO + 16 * j);
#pragma unroll
    for (int i = 0; i < 4; ++i)
#pragma unroll
      for (int j = 0; j < 4; ++j) c[i][j] = wmma_bf16(a[i], b[j], c[i][j]);
  }
#pragma unroll
  for (int i = 0; i < 4; ++i)
#pragma unroll
    for (int j = 0; j < 4; ++j)
#pragma unroll
      for (int r = 0; r < 8; ++r) {
        int l = tileL + 16 * i + r + 8 * half;
        int o = tileO + 16 * j + n;
        int hh = o >> 6, hd = o & 63;
        dst[((size_t)hh * L_SEQ + l) * HD + hd] = (__bf16)c[i][j][r];
      }
}

// K1 fallback: f32 operands, inline conversion, 32x32 tiles.
__global__ __launch_bounds__(32)
void k_proj_qkv(const float* __restrict__ hs,
                const float* __restrict__ Wq, const float* __restrict__ Wk,
                const float* __restrict__ Wv,
                __bf16* __restrict__ qw, __bf16* __restrict__ kw,
                __bf16* __restrict__ vw) {
  const int tileL = blockIdx.x * 32;
  const int tileO = blockIdx.y * 32;
  const int z = blockIdx.z;
  const float* W = (z == 0) ? Wq : (z == 1) ? Wk : Wv;
  __bf16* dst    = (z == 0) ? qw : (z == 1) ? kw : vw;
  const int lane = threadIdx.x & 31;
  const int n = lane & 15, half = lane >> 4;

  v8f c[2][2] = {};
  for (int k0 = 0; k0 < HID; k0 += 32) {
    v16bf a0 = load_a_from_f32(hs, HID, tileL, k0);
    v16bf a1 = load_a_from_f32(hs, HID, tileL + 16, k0);
    v16bf b0 = load_b_wT_f32(W, HID, k0, tileO);
    v16bf b1 = load_b_wT_f32(W, HID, k0, tileO + 16);
    c[0][0] = wmma_bf16(a0, b0, c[0][0]);
    c[0][1] = wmma_bf16(a0, b1, c[0][1]);
    c[1][0] = wmma_bf16(a1, b0, c[1][0]);
    c[1][1] = wmma_bf16(a1, b1, c[1][1]);
  }
#pragma unroll
  for (int i = 0; i < 2; ++i)
#pragma unroll
    for (int j = 0; j < 2; ++j)
#pragma unroll
      for (int r = 0; r < 8; ++r) {
        int l = tileL + 16 * i + r + 8 * half;
        int o = tileO + 16 * j + n;
        int hh = o >> 6, hd = o & 63;
        dst[((size_t)hh * L_SEQ + l) * HD + hd] = (__bf16)c[i][j][r];
      }
}

// ---------------------------------------------------------------------------
// K2: fq/fk = softmax( x @ Wf[h] ) over the 64-wide feature dim.
// ---------------------------------------------------------------------------
__global__ __launch_bounds__(32)
void k_feature(const __bf16* __restrict__ qw, const __bf16* __restrict__ kw,
               const float* __restrict__ Wfq, const float* __restrict__ Wfk,
               __bf16* __restrict__ fqw, __bf16* __restrict__ fkw) {
  const int l0 = blockIdx.x * 16;
  const int h  = blockIdx.y;
  const int z  = blockIdx.z;
  const __bf16* Xh = ((z == 0) ? qw : kw) + (size_t)h * L_SEQ * HD;
  const float*  Wh = ((z == 0) ? Wfq : Wfk) + (size_t)h * HD * FD;
  __bf16*       F  = (z == 0) ? fqw : fkw;
  const int lane = threadIdx.x & 31, n = lane & 15, half = lane >> 4;

  v8f c[4] = {};
#pragma unroll
  for (int ks = 0; ks < 2; ++ks) {
    v16bf a = load_a_from_bf16(Xh, HD, l0, ks * 32);
#pragma unroll
    for (int j = 0; j < 4; ++j) {
      v16bf b = load_b_kn_f32(Wh, FD, ks * 32, j * 16);
      c[j] = wmma_bf16(a, b, c[j]);
    }
  }
#pragma unroll
  for (int r = 0; r < 8; ++r) {
    float m = fmaxf(fmaxf(c[0][r], c[1][r]), fmaxf(c[2][r], c[3][r]));
#pragma unroll
    for (int msk = 1; msk < 16; msk <<= 1) m = fmaxf(m, __shfl_xor(m, msk, 32));
    float ex[4];
    float s = 0.0f;
#pragma unroll
    for (int j = 0; j < 4; ++j) { ex[j] = __expf(c[j][r] - m); s += ex[j]; }
#pragma unroll
    for (int msk = 1; msk < 16; msk <<= 1) s += __shfl_xor(s, msk, 32);
    float inv = 1.0f / s;
    int l = l0 + r + 8 * half;
#pragma unroll
    for (int j = 0; j < 4; ++j)
      F[((size_t)h * L_SEQ + l) * FD + j * 16 + n] = (__bf16)(ex[j] * inv);
  }
}

// ---------------------------------------------------------------------------
// K3: chunked causal linear attention (chunk = 32 tokens, one wave per head).
//   Y  = Q·S_prev + tril(Q·K^T)·V_ext      (V_ext ones column -> Y[:,64] is
//   S += K^T·V_ext                          the normalizer q·cumsum(k))
// fk/v chunk tiles staged to LDS with GLOBAL_LOAD_ASYNC_TO_LDS_B128 when the
// toolchain exposes it; S re-fragmented C/D->B via an LDS bounce each chunk.
// ---------------------------------------------------------------------------
__global__ __launch_bounds__(32)
void k_linattn(const __bf16* __restrict__ fqw, const __bf16* __restrict__ fkw,
               const __bf16* __restrict__ vw, __bf16* __restrict__ yw) {
  __shared__ __align__(16) __bf16 Slds[64 * 80];
  __shared__ __align__(16) __bf16 A2lds[32 * 32];
#if HAVE_ASYNC
  __shared__ __align__(16) __bf16 fkC[32 * 64];
  __shared__ __align__(16) __bf16 vC[32 * 64];
#endif
  const int h = blockIdx.x;
  const int lane = threadIdx.x & 31, n = lane & 15, half = lane >> 4;
  const __bf16* fqh = fqw + (size_t)h * L_SEQ * FD;
  const __bf16* fkh = fkw + (size_t)h * L_SEQ * FD;
  const __bf16* vh  = vw  + (size_t)h * L_SEQ * HD;

  v8f S[4][5] = {};              // state [64f x 80d], f32 accumulators
  v16bf vb4;                     // ones-column B fragment (denominator trick)
#pragma unroll
  for (int e = 0; e < 16; ++e) vb4[e] = (n == 0) ? (__bf16)1.0f : (__bf16)0.0f;

  for (int l0 = 0; l0 < L_SEQ; l0 += 32) {
#if HAVE_ASYNC
    {  // kick off async chunk staging; latency hidden behind S staging below
      const __bf16* gk = fkh + (size_t)l0 * FD;
      const __bf16* gv = vh + (size_t)l0 * HD;
#pragma unroll
      for (int it = 0; it < 8; ++it) {
        int eo = (it * 32 + lane) * 8;  // 8 bf16 = 16 bytes per lane
        async_cp16(fkC + eo, gk + eo);
        async_cp16(vC + eo, gv + eo);
      }
    }
    const __bf16* fkc = fkC;
    const __bf16* vc  = vC;
#else
    const __bf16* fkc = fkh + (size_t)l0 * FD;
    const __bf16* vc  = vh + (size_t)l0 * HD;
#endif

    // stage pre-chunk state to LDS as bf16 [f][d] row-major
#pragma unroll
    for (int fi = 0; fi < 4; ++fi)
#pragma unroll
      for (int j = 0; j < 5; ++j)
#pragma unroll
        for (int r = 0; r < 8; ++r)
          Slds[(fi * 16 + r + 8 * half) * 80 + j * 16 + n] = (__bf16)S[fi][j][r];
    __syncthreads();

    // Q fragments (A operands), reused 3x
    v16bf qa[2][2];
#pragma unroll
    for (int i = 0; i < 2; ++i)
#pragma unroll
      for (int ks = 0; ks < 2; ++ks)
        qa[i][ks] = load_a_from_bf16(fqh, FD, l0 + 16 * i, ks * 32);

#if HAVE_ASYNC
    wait_async0();  // fk/v chunk now resident in LDS
#endif

    // A = Q·K^T  (contraction over f=64)
    v8f A[2][2] = {};
#pragma unroll
    for (int ks = 0; ks < 2; ++ks)
#pragma unroll
      for (int j2 = 0; j2 < 2; ++j2) {
        v16bf kb = load_b_nk_bf16(fkc, FD, ks * 32, j2 * 16);
#pragma unroll
        for (int i = 0; i < 2; ++i)
          A[i][j2] = wmma_bf16(qa[i][ks], kb, A[i][j2]);
      }

    // inclusive causal mask, stash bf16 -> LDS, reload as A fragments
#pragma unroll
    for (int i = 0; i < 2; ++i)
#pragma unroll
      for (int j2 = 0; j2 < 2; ++j2)
#pragma unroll
        for (int r = 0; r < 8; ++r) {
          int row = 16 * i + r + 8 * half;
          int col = 16 * j2 + n;
          A2lds[row * 32 + col] = (__bf16)((col <= row) ? A[i][j2][r] : 0.0f);
        }
    __syncthreads();
    v16bf a2[2];
#pragma unroll
    for (int i = 0; i < 2; ++i) a2[i] = load_a_from_bf16(A2lds, 32, 16 * i, 0);

    // V fragments (B operands), reused for Y and for the S update
    v16bf vb[4];
#pragma unroll
    for (int j = 0; j < 4; ++j) vb[j] = load_b_kn_bf16(vc, HD, 0, j * 16);

    // denominator tile (j=4): col 0 = q·cumsum(k)
    float dn[2][8];
#pragma unroll
    for (int i = 0; i < 2; ++i) {
      v8f y4 = {};
#pragma unroll
      for (int ks = 0; ks < 2; ++ks) {
        v16bf sb = load_b_kn_bf16(Slds, 80, ks * 32, 64);
        y4 = wmma_bf16(qa[i][ks], sb, y4);
      }
      y4 = wmma_bf16(a2[i], vb4, y4);
#pragma unroll
      for (int r = 0; r < 8; ++r)
        dn[i][r] = __shfl(y4[r], lane & 16, 32) + 1e-12f;  // bcast from N=0 lane
    }

    // output tiles j=0..3: Y = Q·S_prev + tril(QK^T)·V, then /denom
#pragma unroll
    for (int j = 0; j < 4; ++j) {
      v16bf sb0 = load_b_kn_bf16(Slds, 80, 0, j * 16);
      v16bf sb1 = load_b_kn_bf16(Slds, 80, 32, j * 16);
#pragma unroll
      for (int i = 0; i < 2; ++i) {
        v8f y = {};
        y = wmma_bf16(qa[i][0], sb0, y);
        y = wmma_bf16(qa[i][1], sb1, y);
        y = wmma_bf16(a2[i], vb[j], y);
#pragma unroll
        for (int r = 0; r < 8; ++r) {
          int l = l0 + 16 * i + r + 8 * half;
          yw[(size_t)l * HID + h * HD + j * 16 + n] = (__bf16)(y[r] / dn[i][r]);
        }
      }
    }

    // state update: S += K^T · V_ext  (contraction over the 32 chunk tokens)
    v16bf ka[4];
#pragma unroll
    for (int fi = 0; fi < 4; ++fi) ka[fi] = load_aT_bf16(fkc, FD, fi * 16, 0);
#pragma unroll
    for (int fi = 0; fi < 4; ++fi) {
#pragma unroll
      for (int j = 0; j < 4; ++j) S[fi][j] = wmma_bf16(ka[fi], vb[j], S[fi][j]);
      S[fi][4] = wmma_bf16(ka[fi], vb4, S[fi][4]);
    }
    __syncthreads();
  }
}

// ---------------------------------------------------------------------------
// K4: out = y @ Wo^T (f32 output).  bf16-weight fast path + f32 fallback.
// ---------------------------------------------------------------------------
__global__ __launch_bounds__(32)
void k_outproj_bf16(const __bf16* __restrict__ yw, const __bf16* __restrict__ wob,
                    float* __restrict__ out) {
  const int tileL = blockIdx.x * 64;
  const int tileO = blockIdx.y * 64;
  const int lane = threadIdx.x & 31;
  const int n = lane & 15, half = lane >> 4;

  v8f c[4][4] = {};
  for (int k0 = 0; k0 < HID; k0 += 32) {
    v16bf a[4], b[4];
#pragma unroll
    for (int i = 0; i < 4; ++i) a[i] = load_a_from_bf16(yw, HID, tileL + 16 * i, k0);
#pragma unroll
    for (int j = 0; j < 4; ++j) b[j] = load_b_nk_bf16(wob, HID, k0, tileO + 16 * j);
#pragma unroll
    for (int i = 0; i < 4; ++i)
#pragma unroll
      for (int j = 0; j < 4; ++j) c[i][j] = wmma_bf16(a[i], b[j], c[i][j]);
  }
#pragma unroll
  for (int i = 0; i < 4; ++i)
#pragma unroll
    for (int j = 0; j < 4; ++j)
#pragma unroll
      for (int r = 0; r < 8; ++r) {
        int l = tileL + 16 * i + r + 8 * half;
        int o = tileO + 16 * j + n;
        out[(size_t)l * HID + o] = c[i][j][r];
      }
}

__global__ __launch_bounds__(32)
void k_outproj(const __bf16* __restrict__ yw, const float* __restrict__ Wo,
               float* __restrict__ out) {
  const int tileL = blockIdx.x * 32;
  const int tileO = blockIdx.y * 32;
  const int lane = threadIdx.x & 31;
  const int n = lane & 15, half = lane >> 4;

  v8f c[2][2] = {};
  for (int k0 = 0; k0 < HID; k0 += 32) {
    v16bf a0 = load_a_from_bf16(yw, HID, tileL, k0);
    v16bf a1 = load_a_from_bf16(yw, HID, tileL + 16, k0);
    v16bf b0 = load_b_wT_f32(Wo, HID, k0, tileO);
    v16bf b1 = load_b_wT_f32(Wo, HID, k0, tileO + 16);
    c[0][0] = wmma_bf16(a0, b0, c[0][0]);
    c[0][1] = wmma_bf16(a0, b1, c[0][1]);
    c[1][0] = wmma_bf16(a1, b0, c[1][0]);
    c[1][1] = wmma_bf16(a1, b1, c[1][1]);
  }
#pragma unroll
  for (int i = 0; i < 2; ++i)
#pragma unroll
    for (int j = 0; j < 2; ++j)
#pragma unroll
      for (int r = 0; r < 8; ++r) {
        int l = tileL + 16 * i + r + 8 * half;
        int o = tileO + 16 * j + n;
        out[(size_t)l * HID + o] = c[i][j][r];
      }
}

// ---------------------------------------------------------------------------
extern "C" void kernel_launch(void* const* d_in, const int* in_sizes, int n_in,
                              void* d_out, int out_size, void* d_ws, size_t ws_size,
                              hipStream_t stream) {
  (void)in_sizes; (void)n_in; (void)out_size;
  const float* hs  = (const float*)d_in[0];
  const float* Wq  = (const float*)d_in[1];
  const float* Wk  = (const float*)d_in[2];
  const float* Wv  = (const float*)d_in[3];
  const float* Wo  = (const float*)d_in[4];
  const float* Wfq = (const float*)d_in[5];
  const float* Wfk = (const float*)d_in[6];
  float* out = (float*)d_out;

  const size_t NE = (size_t)NH * L_SEQ * HD;        // 2M elements
  __bf16* qw  = (__bf16*)d_ws;
  __bf16* kw  = qw  + NE;
  __bf16* vw  = kw  + NE;
  __bf16* fqw = vw  + NE;
  __bf16* fkw = fqw + NE;
  __bf16* yw  = fkw + NE;                            // [L_SEQ x HID]
  // fast-path extras: bf16 hidden + 4 bf16 weight matrices
  __bf16* hb  = yw  + NE;
  __bf16* wqb = hb  + NE;
  __bf16* wkb = wqb + 2 * NE;
  __bf16* wvb = wkb + 2 * NE;
  __bf16* wob = wvb + 2 * NE;

  const size_t need = (6 * NE + NE + 8 * NE) * sizeof(__bf16);  // 60 MB
  const bool fast = ws_size >= need;

  if (fast) {
    const int nW8 = (int)(2 * NE / 8), nH8 = (int)(NE / 8);
    k_cvt_f32_bf16<<<dim3((nH8 + 255) / 256), 256, 0, stream>>>(hs, hb, nH8);
    k_cvt_f32_bf16<<<dim3((nW8 + 255) / 256), 256, 0, stream>>>(Wq, wqb, nW8);
    k_cvt_f32_bf16<<<dim3((nW8 + 255) / 256), 256, 0, stream>>>(Wk, wkb, nW8);
    k_cvt_f32_bf16<<<dim3((nW8 + 255) / 256), 256, 0, stream>>>(Wv, wvb, nW8);
    k_cvt_f32_bf16<<<dim3((nW8 + 255) / 256), 256, 0, stream>>>(Wo, wob, nW8);
    k_proj_qkv_bf16<<<dim3(L_SEQ / 64, HID / 64, 3), 32, 0, stream>>>(
        hb, wqb, wkb, wvb, qw, kw, vw);
  } else {
    k_proj_qkv<<<dim3(L_SEQ / 32, HID / 32, 3), 32, 0, stream>>>(hs, Wq, Wk, Wv,
                                                                 qw, kw, vw);
  }

  k_feature<<<dim3(L_SEQ / 16, NH, 2), 32, 0, stream>>>(qw, kw, Wfq, Wfk, fqw, fkw);
  k_linattn<<<dim3(NH), 32, 0, stream>>>(fqw, fkw, vw, yw);

  if (fast) {
    k_outproj_bf16<<<dim3(L_SEQ / 64, HID / 64), 32, 0, stream>>>(yw, wob, out);
  } else {
    k_outproj<<<dim3(L_SEQ / 32, HID / 32), 32, 0, stream>>>(yw, Wo, out);
  }
}
